// NodeGCNNet_9036611191473
// MI455X (gfx1250) — compile-verified, matrix operand order checked by reading, not run on hardware
//
#include <hip/hip_runtime.h>
#include <stdint.h>

#define N_NODES 50000
#define MPAD    50176   // 392 * 128, padded row count for bounds-free WMMA GEMM

typedef __attribute__((ext_vector_type(16))) __bf16 v16bf;
typedef __attribute__((ext_vector_type(8)))  float  v8f;

struct Q32 { uint4 lo, hi; };

// Build a 16x32 (or 32x16) bf16 WMMA fragment for one lane: two contiguous
// 16-byte chunks from one row (ISA 7.12.2: elements 0..7 = K base..base+7,
// elements 8..15 = K base+16..base+23).
static __device__ __forceinline__ v16bf make_frag(const unsigned short* p) {
  Q32 q;
  q.lo = *(const uint4*)(p);
  q.hi = *(const uint4*)(p + 16);
  return __builtin_bit_cast(v16bf, q);
}

static __device__ __forceinline__ unsigned short f2bf(float f) {
  unsigned int u = __float_as_uint(f);
  u += 0x7FFFu + ((u >> 16) & 1u);   // round-to-nearest-even
  return (unsigned short)(u >> 16);
}

static __device__ __forceinline__ void atomicAddF32(float* p, float v) {
  __hip_atomic_fetch_add(p, v, __ATOMIC_RELAXED, __HIP_MEMORY_SCOPE_AGENT);
}

// ---------------- graph-norm kernels ----------------

__global__ void deg_kernel(const long long* __restrict__ dst, float* __restrict__ deg, int E) {
  int i = blockIdx.x * blockDim.x + threadIdx.x;
  if (i < E) atomicAddF32(&deg[(int)dst[i]], 1.0f);
}

__global__ void dinv_kernel(float* __restrict__ deg) {
  int i = blockIdx.x * blockDim.x + threadIdx.x;
  if (i < N_NODES) deg[i] = rsqrtf(deg[i] + 1.0f);   // +1 = self loop; always > 0
}

__global__ void norm_kernel(const long long* __restrict__ src, const long long* __restrict__ dst,
                            const float* __restrict__ dinv, float* __restrict__ norm, int E) {
  int i = blockIdx.x * blockDim.x + threadIdx.x;
  if (i < E) norm[i] = dinv[(int)src[i]] * dinv[(int)dst[i]];
}

// ---------------- conversion kernels (8 elems / thread, b128 stores) ----------------

__global__ void cvt_x_kernel(const float* __restrict__ X, unsigned short* __restrict__ Xb,
                             int kLog2, long long total8) {   // total8 = MPAD*K/8
  long long i = (long long)blockIdx.x * blockDim.x + threadIdx.x;
  if (i >= total8) return;
  long long base = i * 8;
  int row = (int)(base >> kLog2);   // K % 8 == 0 -> all 8 elems same row
  unsigned short us[8];
  if (row < N_NODES) {
    float4 a = *(const float4*)(X + base);
    float4 b = *(const float4*)(X + base + 4);
    us[0] = f2bf(a.x); us[1] = f2bf(a.y); us[2] = f2bf(a.z); us[3] = f2bf(a.w);
    us[4] = f2bf(b.x); us[5] = f2bf(b.y); us[6] = f2bf(b.z); us[7] = f2bf(b.w);
  } else {
    #pragma unroll
    for (int j = 0; j < 8; ++j) us[j] = 0;
  }
  *(uint4*)(Xb + base) = *(const uint4*)us;
}

__global__ void cvt_w_kernel(const float* __restrict__ W, unsigned short* __restrict__ Wb,
                             long long total8) {
  long long i = (long long)blockIdx.x * blockDim.x + threadIdx.x;
  if (i >= total8) return;
  long long base = i * 8;
  float4 a = *(const float4*)(W + base);
  float4 b = *(const float4*)(W + base + 4);
  unsigned short us[8];
  us[0] = f2bf(a.x); us[1] = f2bf(a.y); us[2] = f2bf(a.z); us[3] = f2bf(a.w);
  us[4] = f2bf(b.x); us[5] = f2bf(b.y); us[6] = f2bf(b.z); us[7] = f2bf(b.w);
  *(uint4*)(Wb + base) = *(const uint4*)us;
}

// ---------------- WMMA GEMM: C[M,N] = A[M,K] * B[N,K]^T (+bias) ----------------
// A: bf16 row-major [>= 128*grid.x, K], K % 32 == 0
// B: bf16 row-major [N, K] (torch Linear weight layout), N % 128 == 0 or N == grid.y*128
// Block = 256 threads = 8 waves arranged 4(M) x 2(N); block tile 128x128,
// wave tile 32x64 = 2x4 tiles of v_wmma_f32_16x16x32_bf16.
__global__ __launch_bounds__(256) void gemm_bf16_wmma(
    const unsigned short* __restrict__ A,
    const unsigned short* __restrict__ B,
    const float* __restrict__ bias,       // nullptr -> no bias
    float* __restrict__ C,
    int K, int N, int Mstore, int doRelu)
{
  const int lane = threadIdx.x & 31;
  const int wave = threadIdx.x >> 5;
  const int m0   = blockIdx.x * 128 + (wave & 3) * 32;
  const int n0   = blockIdx.y * 128 + (wave >> 2) * 64;
  const int g    = lane >> 4;    // 0/1 : K-half select
  const int r    = lane & 15;    // row-in-tile / col-in-tile

  v8f acc[2][4];
  #pragma unroll
  for (int i = 0; i < 2; ++i)
    #pragma unroll
    for (int t = 0; t < 4; ++t) { v8f z = {}; acc[i][t] = z; }

  const unsigned short* pa0 = A + (size_t)(m0 + r) * K + g * 8;
  const unsigned short* pa1 = pa0 + (size_t)16 * K;
  const unsigned short* pb0 = B + (size_t)(n0 + r) * K + g * 8;

  for (int kk = 0; kk < K; kk += 32) {
    v16bf a0 = make_frag(pa0 + kk);
    v16bf a1 = make_frag(pa1 + kk);
    v16bf b0 = make_frag(pb0 + kk);
    v16bf b1 = make_frag(pb0 + (size_t)16 * K + kk);
    v16bf b2 = make_frag(pb0 + (size_t)32 * K + kk);
    v16bf b3 = make_frag(pb0 + (size_t)48 * K + kk);

    acc[0][0] = __builtin_amdgcn_wmma_f32_16x16x32_bf16(false, a0, false, b0, (short)0, acc[0][0], false, false);
    acc[0][1] = __builtin_amdgcn_wmma_f32_16x16x32_bf16(false, a0, false, b1, (short)0, acc[0][1], false, false);
    acc[0][2] = __builtin_amdgcn_wmma_f32_16x16x32_bf16(false, a0, false, b2, (short)0, acc[0][2], false, false);
    acc[0][3] = __builtin_amdgcn_wmma_f32_16x16x32_bf16(false, a0, false, b3, (short)0, acc[0][3], false, false);
    acc[1][0] = __builtin_amdgcn_wmma_f32_16x16x32_bf16(false, a1, false, b0, (short)0, acc[1][0], false, false);
    acc[1][1] = __builtin_amdgcn_wmma_f32_16x16x32_bf16(false, a1, false, b1, (short)0, acc[1][1], false, false);
    acc[1][2] = __builtin_amdgcn_wmma_f32_16x16x32_bf16(false, a1, false, b2, (short)0, acc[1][2], false, false);
    acc[1][3] = __builtin_amdgcn_wmma_f32_16x16x32_bf16(false, a1, false, b3, (short)0, acc[1][3], false, false);
  }

  // Epilogue: C/D layout (ISA 7.12.2): lanes 0-15 -> M = e, lanes 16-31 -> M = 8+e; N = lane%16.
  #pragma unroll
  for (int i = 0; i < 2; ++i) {
    #pragma unroll
    for (int t = 0; t < 4; ++t) {
      const int n = n0 + t * 16 + r;
      const float bv = bias ? bias[n] : 0.0f;
      const int rowbase = m0 + i * 16 + g * 8;
      #pragma unroll
      for (int e = 0; e < 8; ++e) {
        const int row = rowbase + e;
        if (row < Mstore) {
          float v = acc[i][t][e] + bv;
          if (doRelu) v = fmaxf(v, 0.0f);
          C[(size_t)row * N + n] = v;
        }
      }
    }
  }
}

// ---------------- message scatter: out[dst] += H[src] * norm ----------------
// 4 channels per thread: float4 read of the source row, 4 hw f32 atomics.
__global__ void scatter_kernel(const float* __restrict__ H,
                               const long long* __restrict__ src,
                               const long long* __restrict__ dst,
                               const float* __restrict__ norm,
                               float* __restrict__ out, int E, int cLog2) {
  const int c4Log2 = cLog2 - 2;                       // quads per row
  long long i = (long long)blockIdx.x * blockDim.x + threadIdx.x;
  long long total = (long long)E << c4Log2;
  if (i >= total) return;
  int e  = (int)(i >> c4Log2);
  int c4 = (int)(i & ((1 << c4Log2) - 1));
  int s  = (int)src[e];
  int d  = (int)dst[e];
  float nv = norm[e];
  float4 h = *(const float4*)(H + ((size_t)s << cLog2) + c4 * 4);
  float* o = out + ((size_t)d << cLog2) + c4 * 4;
  atomicAddF32(o + 0, h.x * nv);
  atomicAddF32(o + 1, h.y * nv);
  atomicAddF32(o + 2, h.z * nv);
  atomicAddF32(o + 3, h.w * nv);
}

// ---------------- self-loop + bias + relu (4 elems / thread) ----------------
__global__ void finish_kernel(float* __restrict__ agg, const float* __restrict__ H,
                              const float* __restrict__ dinv, const float* __restrict__ bias,
                              int cLog2, long long total4) {   // total4 = N_NODES*C/4
  long long i = (long long)blockIdx.x * blockDim.x + threadIdx.x;
  if (i >= total4) return;
  long long base = i * 4;
  int row = (int)(base >> cLog2);
  int c   = (int)(base & ((1 << cLog2) - 1));
  float dv = dinv[row];
  float d2 = dv * dv;
  float4 a = *(const float4*)(agg + base);
  float4 h = *(const float4*)(H + base);
  float4 b = *(const float4*)(bias + c);
  float4 o;
  o.x = fmaxf(a.x + h.x * d2 + b.x, 0.0f);
  o.y = fmaxf(a.y + h.y * d2 + b.y, 0.0f);
  o.z = fmaxf(a.z + h.z * d2 + b.z, 0.0f);
  o.w = fmaxf(a.w + h.w * d2 + b.w, 0.0f);
  *(float4*)(agg + base) = o;
}

// ---------------- host ----------------

extern "C" void kernel_launch(void* const* d_in, const int* in_sizes, int n_in,
                              void* d_out, int out_size, void* d_ws, size_t ws_size,
                              hipStream_t stream) {
  const long long* ei   = (const long long*)d_in[0];
  const float* node = (const float*)d_in[1];
  const float* W1 = (const float*)d_in[2]; const float* b1 = (const float*)d_in[3];
  const float* W2 = (const float*)d_in[4]; const float* b2 = (const float*)d_in[5];
  const float* W3 = (const float*)d_in[6]; const float* b3 = (const float*)d_in[7];
  const float* Wfc = (const float*)d_in[8]; const float* bfc = (const float*)d_in[9];

  const int E = in_sizes[0] / 2;
  const long long* srcI = ei;
  const long long* dstI = ei + E;

  // workspace carve-up
  char* w = (char*)d_ws;
  float* dinv = (float*)w;  w += (size_t)50048 * 4;
  float* norm = (float*)w;  w += (size_t)450560 * 4;
  float* Hbuf = (float*)w;  w += (size_t)MPAD * 512 * 4;
  float* Abuf = (float*)w;  w += (size_t)MPAD * 512 * 4;
  unsigned short* Xbf = (unsigned short*)w;  w += (size_t)MPAD * 512 * 2;
  unsigned short* Wbf = (unsigned short*)w;  w += (size_t)1024 * 512 * 2;

  // degrees + symmetric norm
  hipMemsetAsync(dinv, 0, (size_t)N_NODES * 4, stream);
  deg_kernel <<<(E + 255) / 256, 256, 0, stream>>>(dstI, dinv, E);
  dinv_kernel<<<(N_NODES + 255) / 256, 256, 0, stream>>>(dinv);
  norm_kernel<<<(E + 255) / 256, 256, 0, stream>>>(srcI, dstI, dinv, norm, E);

  auto gcn_layer = [&](const float* Xin, const float* Wt, const float* bias,
                       int K, int kLog2, int Cout, int cLog2) {
    long long tX8 = (long long)MPAD * K / 8;
    cvt_x_kernel<<<(unsigned)((tX8 + 255) / 256), 256, 0, stream>>>(Xin, Xbf, kLog2, tX8);
    long long tW8 = (long long)Cout * K / 8;
    cvt_w_kernel<<<(unsigned)((tW8 + 255) / 256), 256, 0, stream>>>(Wt, Wbf, tW8);
    dim3 gg(MPAD / 128, (Cout + 127) / 128);
    gemm_bf16_wmma<<<gg, 256, 0, stream>>>(Xbf, Wbf, nullptr, Hbuf, K, Cout, MPAD, 0);
    hipMemsetAsync(Abuf, 0, (size_t)N_NODES * Cout * 4, stream);
    long long tS = (long long)E << (cLog2 - 2);
    scatter_kernel<<<(unsigned)((tS + 255) / 256), 256, 0, stream>>>(Hbuf, srcI, dstI, norm, Abuf, E, cLog2);
    long long tF4 = ((long long)N_NODES << cLog2) / 4;
    finish_kernel<<<(unsigned)((tF4 + 255) / 256), 256, 0, stream>>>(Abuf, Hbuf, dinv, bias, cLog2, tF4);
  };

  gcn_layer(node, W1, b1, 128, 7, 128, 7);   // layer 1: 128 -> 128
  gcn_layer(Abuf, W2, b2, 128, 7, 256, 8);   // layer 2: 128 -> 256
  gcn_layer(Abuf, W3, b3, 256, 8, 512, 9);   // layer 3: 256 -> 512

  // FC: 512 -> 1024, bias, no relu, store exactly N_NODES rows into d_out
  {
    long long tX8 = (long long)MPAD * 512 / 8;
    cvt_x_kernel<<<(unsigned)((tX8 + 255) / 256), 256, 0, stream>>>(Abuf, Xbf, 9, tX8);
    long long tW8 = (long long)1024 * 512 / 8;
    cvt_w_kernel<<<(unsigned)((tW8 + 255) / 256), 256, 0, stream>>>(Wfc, Wbf, tW8);
    dim3 gg(MPAD / 128, 1024 / 128);
    gemm_bf16_wmma<<<gg, 256, 0, stream>>>(Xbf, Wbf, bfc, (float*)d_out, 512, 1024, N_NODES, 0);
  }
}
